// EncoderBlock_9560597201569
// MI455X (gfx1250) — matrix operand order repaired and checked by reading, not compiled, not run on hardware
//
#include <hip/hip_runtime.h>

#define Bn   4
#define Sn   2048
#define Dn   1024
#define Hnum 16
#define DKn  64
#define DFFn 4096
#define Mrows (Bn * Sn)

typedef __attribute__((ext_vector_type(16))) __bf16 v16bf;
typedef __attribute__((ext_vector_type(8)))  __bf16 v8bf;
typedef __attribute__((ext_vector_type(8)))  float  v8f;

union Frag {
    v16bf v;
    struct { v8bf lo, hi; } h;
    unsigned u[8];
};

__device__ __forceinline__ unsigned short f2bf(float f) {
    union { float f; unsigned u; } c; c.f = f;
    unsigned u = c.u;
    u += 0x7FFFu + ((u >> 16) & 1u);   // round-to-nearest-even
    return (unsigned short)(u >> 16);
}
__device__ __forceinline__ unsigned pack_bf2(float a, float b) {
    return (unsigned)f2bf(a) | ((unsigned)f2bf(b) << 16);
}
__device__ __forceinline__ v8bf ld8(const unsigned short* p) {
    return *reinterpret_cast<const v8bf*>(p);
}
__device__ __forceinline__ v8f wmma_bf16(const Frag& a, const Frag& b, v8f c) {
    return __builtin_amdgcn_wmma_f32_16x16x32_bf16(
        false, a.v, false, b.v, (short)0, c, false, false);
}

// ---- CDNA5 async global->LDS copy (16 B per lane), ASYNCcnt tracked ------
__device__ __forceinline__ void async_copy16(unsigned lds_byte_off, const void* gptr) {
    unsigned long long ga = (unsigned long long)gptr;
    asm volatile("global_load_async_to_lds_b128 %0, %1, off"
                 :: "v"(lds_byte_off), "v"(ga) : "memory");
}
__device__ __forceinline__ void wait_async(int keep4) {
    if (keep4) asm volatile("s_wait_asynccnt 0x4" ::: "memory");
    else       asm volatile("s_wait_asynccnt 0x0" ::: "memory");
}
__device__ __forceinline__ unsigned lds_off(const void* p) {
    return (unsigned)(unsigned long long)p;   // low 32 bits = LDS byte offset
}

// -------------------------------------------------------------------------
// f32 -> bf16 conversion (grid-stride)
// -------------------------------------------------------------------------
__global__ void f32_to_bf16_kernel(const float* __restrict__ in,
                                   unsigned short* __restrict__ out, size_t n) {
    for (size_t i = (size_t)blockIdx.x * blockDim.x + threadIdx.x; i < n;
         i += (size_t)gridDim.x * blockDim.x)
        out[i] = f2bf(in[i]);
}

// -------------------------------------------------------------------------
// GEMM: C[M,N] = A[M,K](bf16) * W[N,K](bf16)^T + bias
// MODE 0: f32 out.  MODE 1: bf16 out (+optional ReLU).  MODE 2: Vt scatter.
// 128x128 block tile, BK=32, 8 waves x (32x64), async double-buffered LDS.
// -------------------------------------------------------------------------
#define BM 128
#define BN 128
#define BK 32
#define LDSTR 56   // padded K-pitch (112 B): conflict-free, 16 B aligned

template <int MODE, bool RELU>
__global__ __launch_bounds__(256) void gemm_bf16(
    const unsigned short* __restrict__ A,
    const unsigned short* __restrict__ W,
    const float* __restrict__ bias,
    void* __restrict__ Cout,
    int M, int N, int K)
{
    __shared__ unsigned short sA[2][BM * LDSTR];
    __shared__ unsigned short sB[2][BN * LDSTR];

    const int tid  = threadIdx.x;
    const int wave = tid >> 5;
    const int lane = tid & 31;
    const int lr   = lane & 15;
    const int lh   = lane >> 4;

    const int m0 = blockIdx.x * BM;
    const int n0 = blockIdx.y * BN;
    const int wm = (wave >> 1) * 32;
    const int wn = (wave & 1) * 64;

    // per-thread staging coordinates: 2 x 16B for A, 2 x 16B for W
    int r0 = (tid * 8) >> 5,          c0 = (tid * 8) & 31;
    int r1 = (tid * 8 + 2048) >> 5,   c1 = (tid * 8 + 2048) & 31;

    v8f acc[2][4];
#pragma unroll
    for (int i = 0; i < 2; ++i)
#pragma unroll
        for (int j = 0; j < 4; ++j)
#pragma unroll
            for (int v = 0; v < 8; ++v) acc[i][j][v] = 0.f;

    auto stage = [&](int buf, int k0) {
        async_copy16(lds_off(&sA[buf][r0 * LDSTR + c0]),
                     &A[(size_t)(m0 + r0) * K + k0 + c0]);
        async_copy16(lds_off(&sB[buf][r0 * LDSTR + c0]),
                     &W[(size_t)(n0 + r0) * K + k0 + c0]);
        async_copy16(lds_off(&sA[buf][r1 * LDSTR + c1]),
                     &A[(size_t)(m0 + r1) * K + k0 + c1]);
        async_copy16(lds_off(&sB[buf][r1 * LDSTR + c1]),
                     &W[(size_t)(n0 + r1) * K + k0 + c1]);
    };

    const int T = K / BK;
    stage(0, 0);
    for (int kt = 0; kt < T; ++kt) {
        const int cur = kt & 1;
        const int more = (kt + 1 < T);
        if (more) stage(cur ^ 1, (kt + 1) * BK);
        wait_async(more);
        __syncthreads();

        Frag af[2], bf[4];
#pragma unroll
        for (int i = 0; i < 2; ++i) {   // A layout: K in {8h..8h+7} U {16+8h..}
            const unsigned short* p = &sA[cur][(wm + i * 16 + lr) * LDSTR + lh * 8];
            af[i].h.lo = *reinterpret_cast<const v8bf*>(p);
            af[i].h.hi = *reinterpret_cast<const v8bf*>(p + 16);
        }
#pragma unroll
        for (int j = 0; j < 4; ++j) {   // B layout: K in {8h..8h+7} U {16+8h..}
            const unsigned short* p = &sB[cur][(wn + j * 16 + lr) * LDSTR + lh * 8];
            bf[j].h.lo = *reinterpret_cast<const v8bf*>(p);
            bf[j].h.hi = *reinterpret_cast<const v8bf*>(p + 16);
        }
#pragma unroll
        for (int i = 0; i < 2; ++i)
#pragma unroll
            for (int j = 0; j < 4; ++j)
                acc[i][j] = wmma_bf16(af[i], bf[j], acc[i][j]);
        __syncthreads();   // all waves done with `cur` before it is re-staged
    }

    // Epilogue.  C layout: n = lane%16, m = 8*(lane/16) + v
#pragma unroll
    for (int i = 0; i < 2; ++i) {
#pragma unroll
        for (int j = 0; j < 4; ++j) {
            int n = n0 + wn + j * 16 + lr;
            float bv = bias ? bias[n] : 0.f;
#pragma unroll
            for (int v = 0; v < 8; ++v) {
                int m = m0 + wm + i * 16 + lh * 8 + v;
                float val = acc[i][j][v] + bv;
                if (RELU) val = val > 0.f ? val : 0.f;
                if (MODE == 0) {
                    ((float*)Cout)[(size_t)m * N + n] = val;
                } else if (MODE == 1) {
                    ((unsigned short*)Cout)[(size_t)m * N + n] = f2bf(val);
                } else {  // V transposed: (B, H, DK, S)
                    int b = m / Sn, s = m % Sn;
                    int hh = n / DKn, d = n % DKn;
                    ((unsigned short*)Cout)
                        [(((size_t)b * Hnum + hh) * DKn + d) * Sn + s] = f2bf(val);
                }
            }
        }
    }
}

// -------------------------------------------------------------------------
// Flash attention: 4 waves/block, same (b,h), consecutive q-blocks.
// K/V chunks staged once per block into LDS via async copies (double buf).
// scores^T = K*Q^T (q on lane axis); online softmax; ctx^T = Vt*P^T.
// -------------------------------------------------------------------------
#define KPITCH 72   // 32x64 K chunk, padded (144 B rows)
#define VPITCH 56   // 64x32 V chunk, padded (112 B rows)

__global__ __launch_bounds__(128) void attn_kernel(
    const unsigned short* __restrict__ Q,    // (B*S, D) bf16
    const unsigned short* __restrict__ Km,   // (B*S, D) bf16
    const unsigned short* __restrict__ Vt,   // (B, H, DK, S) bf16
    unsigned short* __restrict__ ctx)        // (B*S, D) bf16
{
    __shared__ unsigned short sK[2][32 * KPITCH];
    __shared__ unsigned short sV[2][64 * VPITCH];

    const float scale = 0.125f;  // 1/sqrt(64)
    const int tid  = threadIdx.x;
    const int wave = tid >> 5;
    const int lane = tid & 31;
    const int lr = lane & 15, lh = lane >> 4;

    const int bh = blockIdx.x >> 5;             // 32 blocks per (b,h)
    const int qb = (blockIdx.x & 31) * 4 + wave;
    const int h = bh % Hnum;
    const int b = bh / Hnum;

    const unsigned short* Qbase = Q + (size_t)(b * Sn) * Dn + h * DKn;
    const unsigned short* Kbase = Km + (size_t)(b * Sn) * Dn + h * DKn;
    const unsigned short* Vbase = Vt + ((size_t)(b * Hnum + h)) * DKn * Sn;

    // staging coords: K chunk 32x64 (16 elems/thread), V chunk 64x32
    const int krow = tid >> 2, kcol = (tid & 3) * 16;
    const int vrow = tid >> 1, vcol = (tid & 1) * 16;

    auto stage = [&](int buf, int kb) {
        async_copy16(lds_off(&sK[buf][krow * KPITCH + kcol]),
                     Kbase + (size_t)(kb + krow) * Dn + kcol);
        async_copy16(lds_off(&sK[buf][krow * KPITCH + kcol + 8]),
                     Kbase + (size_t)(kb + krow) * Dn + kcol + 8);
        async_copy16(lds_off(&sV[buf][vrow * VPITCH + vcol]),
                     Vbase + (size_t)vrow * Sn + kb + vcol);
        async_copy16(lds_off(&sV[buf][vrow * VPITCH + vcol + 8]),
                     Vbase + (size_t)vrow * Sn + kb + vcol + 8);
    };

    // Q as B-operand of K*Q^T: lane col q = lr, k in {8h..8h+7} U {16+8h..} per slice
    Frag qf[2];
    {
        const unsigned short* qr = Qbase + (size_t)(qb * 16 + lr) * Dn + lh * 8;
        qf[0].h.lo = ld8(qr);      qf[0].h.hi = ld8(qr + 16);
        qf[1].h.lo = ld8(qr + 32); qf[1].h.hi = ld8(qr + 48);
    }

    float m_run = -3e38f, l_run = 0.f;
    v8f cacc[4];
#pragma unroll
    for (int t = 0; t < 4; ++t)
#pragma unroll
        for (int v = 0; v < 8; ++v) cacc[t][v] = 0.f;

    stage(0, 0);
    for (int it = 0; it < Sn / 32; ++it) {
        const int kb = it * 32;
        const int cur = it & 1;
        const int more = (kb + 32 < Sn);
        if (more) stage(cur ^ 1, kb + 32);
        wait_async(more);
        __syncthreads();

        // K rows as A-operand (two 16-key subtiles x two 32-wide dk slices)
        Frag ka[2][2];
#pragma unroll
        for (int t = 0; t < 2; ++t) {
            const unsigned short* kr = &sK[cur][(t * 16 + lr) * KPITCH];
            ka[t][0].h.lo = *reinterpret_cast<const v8bf*>(kr + lh * 8);
            ka[t][0].h.hi = *reinterpret_cast<const v8bf*>(kr + 16 + lh * 8);
            ka[t][1].h.lo = *reinterpret_cast<const v8bf*>(kr + 32 + lh * 8);
            ka[t][1].h.hi = *reinterpret_cast<const v8bf*>(kr + 48 + lh * 8);
        }
        v8f st0, st1;
#pragma unroll
        for (int v = 0; v < 8; ++v) { st0[v] = 0.f; st1[v] = 0.f; }
        st0 = wmma_bf16(ka[0][0], qf[0], st0);
        st0 = wmma_bf16(ka[0][1], qf[1], st0);
        st1 = wmma_bf16(ka[1][0], qf[0], st1);
        st1 = wmma_bf16(ka[1][1], qf[1], st1);

        // online softmax over this 32-key chunk (per-q = per-lane)
        float mx = -3e38f;
#pragma unroll
        for (int v = 0; v < 8; ++v) mx = fmaxf(mx, fmaxf(st0[v], st1[v]));
        mx = fmaxf(mx, __shfl_xor(mx, 16));
        float m_new = fmaxf(m_run, mx);
        float corr = __expf(scale * (m_run - m_new));
        float e0[8], e1[8], ps = 0.f;
#pragma unroll
        for (int v = 0; v < 8; ++v) {
            e0[v] = __expf(scale * (st0[v] - m_new));
            e1[v] = __expf(scale * (st1[v] - m_new));
            ps += e0[v] + e1[v];
        }
        ps += __shfl_xor(ps, 16);
        l_run = l_run * corr + ps;
        m_run = m_new;
#pragma unroll
        for (int t = 0; t < 4; ++t)
#pragma unroll
            for (int v = 0; v < 8; ++v) cacc[t][v] *= corr;

        // repack P (C layout) -> B-operand fragment of P^T (lane-16 swap)
        float fa[8], fb[8];
#pragma unroll
        for (int v = 0; v < 8; ++v) {
            float o0 = __shfl_xor(e0[v], 16);
            float o1 = __shfl_xor(e1[v], 16);
            fa[v] = lh ? o1 : e0[v];   // B half-0 needs k 0..15
            fb[v] = lh ? e1[v] : o0;   // B half-1 needs k 16..31
        }
        Frag pb;
#pragma unroll
        for (int j = 0; j < 4; ++j) {
            pb.u[j]     = pack_bf2(fa[2 * j], fa[2 * j + 1]);
            pb.u[4 + j] = pack_bf2(fb[2 * j], fb[2 * j + 1]);
        }

        // ctx^T += Vt_chunk (A) * P^T (B)
#pragma unroll
        for (int t = 0; t < 4; ++t) {
            const unsigned short* vr = &sV[cur][(t * 16 + lr) * VPITCH];
            Frag va;
            va.h.lo = *reinterpret_cast<const v8bf*>(vr + lh * 16);
            va.h.hi = *reinterpret_cast<const v8bf*>(vr + lh * 16 + 8);
            cacc[t] = wmma_bf16(va, pb, cacc[t]);
        }
        __syncthreads();   // all waves done with `cur` before it is re-staged
    }

    float inv = 1.f / l_run;
    const int q = qb * 16 + lr;  // C layout: lane = q column
#pragma unroll
    for (int t = 0; t < 4; ++t)
#pragma unroll
        for (int v = 0; v < 8; ++v) {
            int d = t * 16 + lh * 8 + v;
            ctx[(size_t)(b * Sn + q) * Dn + h * DKn + d] = f2bf(cacc[t][v] * inv);
        }
}

// -------------------------------------------------------------------------
// out = resid + LayerNorm(sub) * g + beta ; one 256-thread block per row
// -------------------------------------------------------------------------
__global__ __launch_bounds__(256) void add_ln_kernel(
    const float* __restrict__ resid, const float* __restrict__ sub,
    const float* __restrict__ g, const float* __restrict__ be,
    float* __restrict__ outf, unsigned short* __restrict__ outb,
    int D, int writeb)
{
    __shared__ float ws[8], ws2[8];
    __shared__ float s_mu, s_rs;
    const int row = blockIdx.x;
    const float* s = sub + (size_t)row * D;
    const float* r = resid + (size_t)row * D;
    float sum = 0.f, sq = 0.f;
    for (int i = threadIdx.x; i < D; i += 256) {
        float v = s[i]; sum += v; sq += v * v;
    }
#pragma unroll
    for (int o = 16; o > 0; o >>= 1) {
        sum += __shfl_xor(sum, o);
        sq  += __shfl_xor(sq, o);
    }
    if ((threadIdx.x & 31) == 0) { ws[threadIdx.x >> 5] = sum; ws2[threadIdx.x >> 5] = sq; }
    __syncthreads();
    if (threadIdx.x == 0) {
        float ts = 0.f, tq = 0.f;
        for (int i = 0; i < 8; ++i) { ts += ws[i]; tq += ws2[i]; }
        float mu = ts / D;
        float var = tq / D - mu * mu;
        s_mu = mu; s_rs = rsqrtf(var + 1e-5f);
    }
    __syncthreads();
    float mu = s_mu, rs = s_rs;
    for (int i = threadIdx.x; i < D; i += 256) {
        float v = (s[i] - mu) * rs * g[i] + be[i] + r[i];
        outf[(size_t)row * D + i] = v;
        if (writeb) outb[(size_t)row * D + i] = f2bf(v);
    }
}

// -------------------------------------------------------------------------
extern "C" void kernel_launch(void* const* d_in, const int* in_sizes, int n_in,
                              void* d_out, int out_size, void* d_ws, size_t ws_size,
                              hipStream_t stream) {
    const float* x   = (const float*)d_in[0];
    const float* wq  = (const float*)d_in[1];
    const float* bq  = (const float*)d_in[2];
    const float* wk  = (const float*)d_in[3];
    const float* bk  = (const float*)d_in[4];
    const float* wv  = (const float*)d_in[5];
    const float* bv  = (const float*)d_in[6];
    const float* wo  = (const float*)d_in[7];
    const float* bo  = (const float*)d_in[8];
    const float* w1  = (const float*)d_in[9];
    const float* b1  = (const float*)d_in[10];
    const float* w2  = (const float*)d_in[11];
    const float* b2  = (const float*)d_in[12];
    const float* g1  = (const float*)d_in[13];
    const float* be1 = (const float*)d_in[14];
    const float* g2  = (const float*)d_in[15];
    const float* be2 = (const float*)d_in[16];

    char* wsb = (char*)d_ws;
    size_t off = 0;
    auto alloc = [&](size_t bytes) -> void* {
        off = (off + 255) & ~(size_t)255;
        void* p = wsb + off;
        off += bytes;
        return p;
    };

    unsigned short* xb   = (unsigned short*)alloc((size_t)Mrows * Dn * 2);
    unsigned short* wqb  = (unsigned short*)alloc((size_t)Dn * Dn * 2);
    unsigned short* wkb  = (unsigned short*)alloc((size_t)Dn * Dn * 2);
    unsigned short* wvb  = (unsigned short*)alloc((size_t)Dn * Dn * 2);
    unsigned short* wob  = (unsigned short*)alloc((size_t)Dn * Dn * 2);
    unsigned short* w1b  = (unsigned short*)alloc((size_t)DFFn * Dn * 2);
    unsigned short* w2b  = (unsigned short*)alloc((size_t)Dn * DFFn * 2);
    unsigned short* Qb   = (unsigned short*)alloc((size_t)Mrows * Dn * 2);
    unsigned short* Kb   = (unsigned short*)alloc((size_t)Mrows * Dn * 2);
    unsigned short* Vtb  = (unsigned short*)alloc((size_t)Mrows * Dn * 2);
    unsigned short* ctxb = (unsigned short*)alloc((size_t)Mrows * Dn * 2);
    float*          attnf= (float*)alloc((size_t)Mrows * Dn * 4);
    float*          x1f  = (float*)alloc((size_t)Mrows * Dn * 4);
    unsigned short* x1b  = (unsigned short*)alloc((size_t)Mrows * Dn * 2);
    unsigned short* h1b  = (unsigned short*)alloc((size_t)Mrows * DFFn * 2);
    float*          ff   = (float*)alloc((size_t)Mrows * Dn * 4);

    auto cvt = [&](const float* src, unsigned short* dst, size_t n) {
        f32_to_bf16_kernel<<<4096, 256, 0, stream>>>(src, dst, n);
    };
    cvt(x,  xb,  (size_t)Mrows * Dn);
    cvt(wq, wqb, (size_t)Dn * Dn);
    cvt(wk, wkb, (size_t)Dn * Dn);
    cvt(wv, wvb, (size_t)Dn * Dn);
    cvt(wo, wob, (size_t)Dn * Dn);
    cvt(w1, w1b, (size_t)DFFn * Dn);
    cvt(w2, w2b, (size_t)Dn * DFFn);

    dim3 blk(256);
    dim3 gD(Mrows / BM, Dn / BN);     // N = 1024
    dim3 gF(Mrows / BM, DFFn / BN);   // N = 4096

    // QKV projections
    gemm_bf16<1, false><<<gD, blk, 0, stream>>>(xb, wqb, bq, Qb, Mrows, Dn, Dn);
    gemm_bf16<1, false><<<gD, blk, 0, stream>>>(xb, wkb, bk, Kb, Mrows, Dn, Dn);
    gemm_bf16<2, false><<<gD, blk, 0, stream>>>(xb, wvb, bv, Vtb, Mrows, Dn, Dn);

    // attention: B*H*(S/16) = 8192 waves, 4 waves/block share K/V in LDS
    attn_kernel<<<2048, 128, 0, stream>>>(Qb, Kb, Vtb, ctxb);

    // output projection + LN1 + residual
    gemm_bf16<0, false><<<gD, blk, 0, stream>>>(ctxb, wob, bo, attnf, Mrows, Dn, Dn);
    add_ln_kernel<<<Mrows, 256, 0, stream>>>(x, attnf, g1, be1, x1f, x1b, Dn, 1);

    // FFN + LN2 + residual
    gemm_bf16<1, true ><<<gF, blk, 0, stream>>>(x1b, w1b, b1, h1b, Mrows, DFFn, Dn);
    gemm_bf16<0, false><<<gD, blk, 0, stream>>>(h1b, w2b, b2, ff, Mrows, Dn, DFFn);
    add_ln_kernel<<<Mrows, 256, 0, stream>>>(x1f, ff, g2, be2, (float*)d_out, nullptr, Dn, 0);
}
